// Denoising_Unet_39857296507352
// MI455X (gfx1250) — compile-verified, matrix operand order checked by reading, not run on hardware
//
#include <hip/hip_runtime.h>
#include <hip/hip_fp16.h>

#define NNODES 50000
#define NEDGES 800000
#define HIDDIM 128
#define NHEADS 4
#define HEADD  32

typedef __attribute__((ext_vector_type(16))) _Float16 v16h;
typedef __attribute__((ext_vector_type(8)))  float    v8f;

__device__ __forceinline__ float gelu_tanh(float x) {
    float x3 = x * x * x;
    return 0.5f * x * (1.0f + tanhf(0.7978845608028654f * (x + 0.044715f * x3)));
}
// monotonic float -> uint encoding for atomic segment-max
__device__ __forceinline__ unsigned enc_f(float f) {
    unsigned u = __float_as_uint(f);
    return (u & 0x80000000u) ? ~u : (u | 0x80000000u);
}
__device__ __forceinline__ float dec_f(unsigned e) {
    unsigned u = (e & 0x80000000u) ? (e & 0x7FFFFFFFu) : ~e;
    return __uint_as_float(u);
}

// ---- cast (+optional concat) f32 -> f16 activations ----
__global__ void k_castcat(const float* __restrict__ s1, const float* __restrict__ s2,
                          _Float16* __restrict__ out, int c1, int c2, int n) {
    int t = blockIdx.x * blockDim.x + threadIdx.x;
    int ct = c1 + c2;
    if (t >= n * ct) return;
    int row = t / ct, col = t - row * ct;
    float v = (col < c1) ? s1[row * c1 + col] : s2[row * c2 + (col - c1)];
    out[t] = (_Float16)v;
}

// ---- weight [K,128] row-major -> WT [128,K] f16 (column-major weight) ----
__global__ void k_transW(const float* __restrict__ W, _Float16* __restrict__ WT, int K) {
    int t = blockIdx.x * blockDim.x + threadIdx.x;
    if (t >= HIDDIM * K) return;
    int n = t / K, k = t - n * K;
    WT[t] = (_Float16)W[k * HIDDIM + n];
}

// ---- WMMA GEMM: C[N,128] = A16[N,K] @ W[K,128] + bias, optional GELU ----
// one wave -> one 16x16 tile; block = 4 waves -> 16x64; grid.y = 2 covers 128 cols
__global__ void k_gemm_wmma(const _Float16* __restrict__ A, const _Float16* __restrict__ BT,
                            const float* __restrict__ bias, float* __restrict__ C,
                            int N, int K, int act) {
    int lane = threadIdx.x & 31;
    int wv   = threadIdx.x >> 5;
    int m0 = blockIdx.x << 4;
    int n0 = (((int)blockIdx.y << 2) | wv) << 4;
    int half = lane >> 4, l15 = lane & 15;

    union Frag { v16h h; unsigned u[8]; };
    Frag fa, fb;
    v8f acc = {};

    const _Float16* arow = A  + (size_t)(m0 + l15) * K;
    const _Float16* brow = BT + (size_t)(n0 + l15) * K;

    for (int k = 0; k < K; k += 32) {
#pragma unroll
        for (int v = 0; v < 8; ++v) {
            // 16-bit A 16x32 layout: lane = M, VGPR v holds K pair
            int kk = ((v & 4) << 2) + (half << 3) + ((v & 3) << 1);
            fa.u[v] = *(const unsigned*)(arow + k + kk);
            fb.u[v] = *(const unsigned*)(brow + k + kk);
        }
        acc = __builtin_amdgcn_wmma_f32_16x16x32_f16(
            /*neg_a=*/false, fa.h, /*neg_b=*/false, fb.h,
            /*c_mod=*/(short)0, acc, /*reuse_a=*/false, /*reuse_b=*/false);
    }

#pragma unroll
    for (int r = 0; r < 8; ++r) {
        int m = m0 + (half << 3) + r;   // C layout: VGPR r, lane half -> row
        int n = n0 + l15;               // lane&15 -> col
        if (m < N) {
            float v = acc[r] + bias[n];
            C[(size_t)m * HIDDIM + n] = (act == 1) ? gelu_tanh(v) : v;
        }
    }
}

// ---- per-(node,head) init: maxenc = enc(-inf), s = 0 ----
__global__ void k_init_ms(unsigned* __restrict__ maxenc, float* __restrict__ s) {
    int t = blockIdx.x * blockDim.x + threadIdx.x;
    if (t >= NNODES * NHEADS) return;
    maxenc[t] = 0x007FFFFFu;  // enc(-inf)
    s[t] = 0.0f;
}

// ---- per-(edge,head): logit = <leaky(fs[src]+fd[dst]), attn[h]>, segment-max ----
__global__ void k_edge_logits(const int* __restrict__ src, const int* __restrict__ dst,
                              const float* __restrict__ fs, const float* __restrict__ fd,
                              const float* __restrict__ attn, float* __restrict__ logits,
                              unsigned* __restrict__ maxenc) {
    int t = blockIdx.x * blockDim.x + threadIdx.x;
    if (t >= NEDGES * NHEADS) return;
    int e = t >> 2, hd = t & 3;
    int si = src[e], di = dst[e];
    const float4* ps = (const float4*)(fs + (size_t)si * HIDDIM + hd * HEADD);
    const float4* pd = (const float4*)(fd + (size_t)di * HIDDIM + hd * HEADD);
    const float4* pa = (const float4*)(attn + hd * HEADD);
    float acc = 0.0f;
#pragma unroll
    for (int i = 0; i < 8; ++i) {
        float4 a = ps[i], b = pd[i], w = pa[i];
        float v;
        v = a.x + b.x; v = (v > 0.0f) ? v : 0.2f * v; acc += v * w.x;
        v = a.y + b.y; v = (v > 0.0f) ? v : 0.2f * v; acc += v * w.y;
        v = a.z + b.z; v = (v > 0.0f) ? v : 0.2f * v; acc += v * w.z;
        v = a.w + b.w; v = (v > 0.0f) ? v : 0.2f * v; acc += v * w.w;
    }
    logits[t] = acc;
    atomicMax(maxenc + (size_t)di * NHEADS + hd, enc_f(acc));
}

// ---- per-(edge,head): p = exp(logit - m[dst]), segment-sum; p overwrites logits ----
__global__ void k_edge_psum(const int* __restrict__ dst, float* __restrict__ logits,
                            const unsigned* __restrict__ maxenc, float* __restrict__ s) {
    int t = blockIdx.x * blockDim.x + threadIdx.x;
    if (t >= NEDGES * NHEADS) return;
    int e = t >> 2, hd = t & 3;
    int di = dst[e];
    float m = dec_f(maxenc[(size_t)di * NHEADS + hd]);
    if ((__float_as_uint(m) & 0x7F800000u) == 0x7F800000u) m = 0.0f;  // inf/nan -> 0
    float p = expf(logits[t] - m);
    logits[t] = p;
    atomicAdd(s + (size_t)di * NHEADS + hd, p);
}

__global__ void k_zero(float* __restrict__ p, int n) {
    int t = blockIdx.x * blockDim.x + threadIdx.x;
    if (t < n) p[t] = 0.0f;
}

// ---- per-(edge,head): out[dst] += (p/s) * fs[src] ----
__global__ void k_edge_aggregate(const int* __restrict__ src, const int* __restrict__ dst,
                                 const float* __restrict__ p, const float* __restrict__ s,
                                 const float* __restrict__ fs, float* __restrict__ out) {
    int t = blockIdx.x * blockDim.x + threadIdx.x;
    if (t >= NEDGES * NHEADS) return;
    int e = t >> 2, hd = t & 3;
    int si = src[e], di = dst[e];
    float alpha = p[t] / (s[(size_t)di * NHEADS + hd] + 1e-9f);
    const float4* pf = (const float4*)(fs + (size_t)si * HIDDIM + hd * HEADD);
    float* po = out + (size_t)di * HIDDIM + hd * HEADD;
#pragma unroll
    for (int i = 0; i < 8; ++i) {
        float4 v = pf[i];
        atomicAdd(po + 4 * i + 0, alpha * v.x);
        atomicAdd(po + 4 * i + 1, alpha * v.y);
        atomicAdd(po + 4 * i + 2, alpha * v.z);
        atomicAdd(po + 4 * i + 3, alpha * v.w);
    }
}

// ---- h = gelu(out + bout) ----
__global__ void k_finalize(const float* __restrict__ out, const float* __restrict__ bout,
                           float* __restrict__ h) {
    int t = blockIdx.x * blockDim.x + threadIdx.x;
    if (t >= NNODES * HIDDIM) return;
    h[t] = gelu_tanh(out[t] + bout[t & (HIDDIM - 1)]);
}

__global__ void k_copyf(const float* __restrict__ a, float* __restrict__ b, int n) {
    int t = blockIdx.x * blockDim.x + threadIdx.x;
    if (t < n) b[t] = a[t];
}

extern "C" void kernel_launch(void* const* d_in, const int* in_sizes, int n_in,
                              void* d_out, int out_size, void* d_ws, size_t ws_size,
                              hipStream_t stream) {
    const float* x_t        = (const float*)d_in[0];
    const float* time_embed = (const float*)d_in[1];
    const int*   src        = (const int*)d_in[2];
    const int*   dst        = (const int*)d_in[3];
    const float* W_init = (const float*)d_in[4];   const float* b_init = (const float*)d_in[5];
    const float* Wsrc_down = (const float*)d_in[6];  const float* bsrc_down = (const float*)d_in[7];
    const float* Wdst_down = (const float*)d_in[8];  const float* bdst_down = (const float*)d_in[9];
    const float* attn_down = (const float*)d_in[10]; const float* bout_down = (const float*)d_in[11];
    const float* Wsrc_mid = (const float*)d_in[12];  const float* bsrc_mid = (const float*)d_in[13];
    const float* Wdst_mid = (const float*)d_in[14];  const float* bdst_mid = (const float*)d_in[15];
    const float* attn_mid = (const float*)d_in[16];  const float* bout_mid = (const float*)d_in[17];
    const float* Wsrc_up = (const float*)d_in[18];   const float* bsrc_up = (const float*)d_in[19];
    const float* Wdst_up = (const float*)d_in[20];   const float* bdst_up = (const float*)d_in[21];
    const float* attn_up = (const float*)d_in[22];   const float* bout_up = (const float*)d_in[23];
    const float* W_fin = (const float*)d_in[24];     const float* b_fin = (const float*)d_in[25];
    float* out = (float*)d_out;

    // workspace carve-up (256B aligned)
    char* ws = (char*)d_ws;
    size_t off = 0;
    auto alloc = [&](size_t bytes) { size_t o = off; off += (bytes + 255) & ~(size_t)255; return o; };
    float*    bufA   = (float*)(ws + alloc((size_t)NNODES * HIDDIM * 4));
    float*    bufB   = (float*)(ws + alloc((size_t)NNODES * HIDDIM * 4));
    float*    bufC   = (float*)(ws + alloc((size_t)NNODES * HIDDIM * 4));
    _Float16* A16    = (_Float16*)(ws + alloc((size_t)NNODES * 256 * 2));
    _Float16* WTa    = (_Float16*)(ws + alloc((size_t)HIDDIM * 256 * 2));
    _Float16* WTb    = (_Float16*)(ws + alloc((size_t)HIDDIM * 256 * 2));
    float*    fs     = (float*)(ws + alloc((size_t)NNODES * HIDDIM * 4));
    float*    fd     = (float*)(ws + alloc((size_t)NNODES * HIDDIM * 4));  // reused as out-accum
    float*    logits = (float*)(ws + alloc((size_t)NEDGES * NHEADS * 4)); // reused as p
    unsigned* maxenc = (unsigned*)(ws + alloc((size_t)NNODES * NHEADS * 4));
    float*    ssum   = (float*)(ws + alloc((size_t)NNODES * NHEADS * 4));

    const int TB = 256;
    auto cdiv = [](long a, long b) { return (int)((a + b - 1) / b); };

    auto gat = [&](const float* hin, const float* hskip,
                   const float* Wsrc, const float* bsrc,
                   const float* Wdst, const float* bdst,
                   const float* attn, const float* bout, float* hout) {
        int K = HIDDIM + (hskip ? HIDDIM : 0);
        k_castcat<<<cdiv((long)NNODES * K, TB), TB, 0, stream>>>(hin, hskip, A16, HIDDIM, K - HIDDIM, NNODES);
        k_transW<<<cdiv((long)HIDDIM * K, TB), TB, 0, stream>>>(Wsrc, WTa, K);
        k_transW<<<cdiv((long)HIDDIM * K, TB), TB, 0, stream>>>(Wdst, WTb, K);
        dim3 gg(NNODES / 16, 2);
        k_gemm_wmma<<<gg, 128, 0, stream>>>(A16, WTa, bsrc, fs, NNODES, K, 0);
        k_gemm_wmma<<<gg, 128, 0, stream>>>(A16, WTb, bdst, fd, NNODES, K, 0);
        k_init_ms<<<cdiv((long)NNODES * NHEADS, TB), TB, 0, stream>>>(maxenc, ssum);
        k_edge_logits<<<cdiv((long)NEDGES * NHEADS, TB), TB, 0, stream>>>(src, dst, fs, fd, attn, logits, maxenc);
        k_edge_psum<<<cdiv((long)NEDGES * NHEADS, TB), TB, 0, stream>>>(dst, logits, maxenc, ssum);
        k_zero<<<cdiv((long)NNODES * HIDDIM, TB), TB, 0, stream>>>(fd, NNODES * HIDDIM);  // fd now free -> accum
        k_edge_aggregate<<<cdiv((long)NEDGES * NHEADS, TB), TB, 0, stream>>>(src, dst, logits, ssum, fs, fd);
        k_finalize<<<cdiv((long)NNODES * HIDDIM, TB), TB, 0, stream>>>(fd, bout, hout);
    };

    // h0 = gelu(concat(x_t, time_embed) @ W_init + b_init) -> bufA
    k_castcat<<<cdiv((long)NNODES * 256, TB), TB, 0, stream>>>(x_t, time_embed, A16, HIDDIM, HIDDIM, NNODES);
    k_transW<<<cdiv((long)HIDDIM * 256, TB), TB, 0, stream>>>(W_init, WTa, 256);
    {
        dim3 gg(NNODES / 16, 2);
        k_gemm_wmma<<<gg, 128, 0, stream>>>(A16, WTa, b_init, bufA, NNODES, 256, 1);
    }

    const int WW = HIDDIM * HIDDIM, WW2 = 2 * HIDDIM * HIDDIM, AH = NHEADS * HEADD;
    // down0: skip0 = bufA
    gat(bufA, nullptr, Wsrc_down, bsrc_down, Wdst_down, bdst_down, attn_down, bout_down, bufB);
    // down1: skip1 = bufB
    gat(bufB, nullptr, Wsrc_down + WW, bsrc_down + HIDDIM, Wdst_down + WW, bdst_down + HIDDIM,
        attn_down + AH, bout_down + HIDDIM, bufC);
    // mid (in-place on bufC)
    gat(bufC, nullptr, Wsrc_mid, bsrc_mid, Wdst_mid, bdst_mid, attn_mid, bout_mid, bufC);
    // up0: concat(h, skip1=bufB)
    gat(bufC, bufB, Wsrc_up, bsrc_up, Wdst_up, bdst_up, attn_up, bout_up, bufC);
    // up1: concat(h, skip0=bufA)
    gat(bufC, bufA, Wsrc_up + WW2, bsrc_up + HIDDIM, Wdst_up + WW2, bdst_up + HIDDIM,
        attn_up + AH, bout_up + HIDDIM, bufC);

    // out = h @ W_fin + b_fin ; second output = h
    k_castcat<<<cdiv((long)NNODES * HIDDIM, TB), TB, 0, stream>>>(bufC, nullptr, A16, HIDDIM, 0, NNODES);
    k_transW<<<cdiv((long)HIDDIM * HIDDIM, TB), TB, 0, stream>>>(W_fin, WTa, HIDDIM);
    {
        dim3 gg(NNODES / 16, 2);
        k_gemm_wmma<<<gg, 128, 0, stream>>>(A16, WTa, b_fin, out, NNODES, HIDDIM, 0);
    }
    k_copyf<<<cdiv((long)NNODES * HIDDIM, TB), TB, 0, stream>>>(bufC, out + (size_t)NNODES * HIDDIM, NNODES * HIDDIM);
}
